// MemorizerPredecoder_24962349925014
// MI455X (gfx1250) — 8-bit matrix operand order repaired and checked by reading, compiled, not run on hardware
//
#include <hip/hip_runtime.h>
#include <hip/hip_bf16.h>
#include <stdint.h>


typedef __attribute__((ext_vector_type(8))) int v8i;

#define L_DIM 256
#define KSTEPS 4
#define TCHUNKS 4

__global__ void pack_rows(const float* __restrict__ src,
                          unsigned char* __restrict__ dst,
                          int* __restrict__ rowsum) {
  __shared__ int red[L_DIM];
  const int row = blockIdx.x;
  const int j   = threadIdx.x;
  const float f = src[(size_t)row * L_DIM + j];
  const int   v = (f > 0.5f) ? 1 : 0;
  dst[(size_t)row * L_DIM + j] = (unsigned char)v;
  red[j] = v;
  __syncthreads();
  #pragma unroll
  for (int s = L_DIM / 2; s > 0; s >>= 1) {
    if (j < s) red[j] += red[j + s];
    __syncthreads();
  }
  if (j == 0) rowsum[row] = red[0];
}

__global__ void init_match(int* __restrict__ match_idx, int n, int sentinel) {
  const int i = blockIdx.x * blockDim.x + threadIdx.x;
  if (i < n) match_idx[i] = sentinel;
}

__global__ void match_gemm(const unsigned char* __restrict__ Ap,
                           const unsigned char* __restrict__ Kp,
                           const int* __restrict__ sumA,
                           const int* __restrict__ sumK,
                           int* __restrict__ match_idx,
                           int nTileM, int nTileT, int sentinel) {
  const int lane = threadIdx.x & 31;
  const int wave = threadIdx.x >> 5;
  const int waveGlobal = blockIdx.x * 8 + wave;

  const int tileM  = waveGlobal / TCHUNKS;
  const int tchunk = waveGlobal % TCHUNKS;
  if (tileM >= nTileM) return;

  const int half = lane >> 4;
  const int m16  = lane & 15;

  v8i a[KSTEPS];
  const unsigned char* abase = Ap + (size_t)(tileM * 16 + m16) * L_DIM;
  #pragma unroll
  for (int ks = 0; ks < KSTEPS; ++ks) {
    #pragma unroll
    for (int v = 0; v < 8; ++v) {
      const int kb = ks * 64 + (v >> 1) * 16 + half * 8 + (v & 1) * 4;
      a[ks][v] = *(const int*)(abase + kb);
    }
  }

  int rowSum[8];
  #pragma unroll
  for (int r = 0; r < 8; ++r)
    rowSum[r] = sumA[tileM * 16 + half * 8 + r];

  int best[8];
  #pragma unroll
  for (int r = 0; r < 8; ++r) best[r] = sentinel;

  auto kstep = [&](const unsigned char* bbase, int ks, v8i c) -> v8i {
    v8i b;
    #pragma unroll
    for (int v = 0; v < 8; ++v) {
      const int kb = ks * 64 + (v >> 1) * 16 + half * 8 + (v & 1) * 4;
      b[v] = *(const int*)(bbase + kb);
    }
    return __builtin_amdgcn_wmma_i32_16x16x64_iu8(
         false, a[ks],  false, b, c,
         false,  false);
  };

  auto epilogue = [&](const v8i& c, int col) {
    const int colSum = sumK[col];
    #pragma unroll
    for (int r = 0; r < 8; ++r) {
      const bool m = (2 * c[r] == rowSum[r] + colSum);
      best[r] = (m && col < best[r]) ? col : best[r];
    }
  };

  const int tilesPerChunk = nTileT / TCHUNKS;
  const int ttBegin = tchunk * tilesPerChunk;

  for (int tt = ttBegin; tt < ttBegin + tilesPerChunk; tt += 2) {
    const unsigned char* bbase0 = Kp + (size_t)((tt + 0) * 16 + m16) * L_DIM;
    const unsigned char* bbase1 = Kp + (size_t)((tt + 1) * 16 + m16) * L_DIM;

    v8i c0 = {};
    #pragma unroll
    for (int ks = 0; ks < KSTEPS; ++ks) c0 = kstep(bbase0, ks, c0);

    v8i c1 = {};
    #pragma unroll
    for (int ks = 0; ks < KSTEPS; ++ks) c1 = kstep(bbase1, ks, c1);

    epilogue(c0, (tt + 0) * 16 + m16);
    epilogue(c1, (tt + 1) * 16 + m16);
  }

  #pragma unroll
  for (int r = 0; r < 8; ++r) {
    if (best[r] < sentinel) {
      const int row = tileM * 16 + half * 8 + r;
      atomicMin(&match_idx[row], best[r]);
    }
  }
}

__global__ void gather_out(const int* __restrict__ match_idx,
                           const float* __restrict__ values,
                           float* __restrict__ out, int T) {
  const int row = blockIdx.x;
  const int j   = threadIdx.x;
  const int idx = match_idx[row];
  float v = 0.0f;
  if (idx < T) v = values[(size_t)idx * L_DIM + j];
  out[(size_t)row * L_DIM + j] = v;
}

extern "C" void kernel_launch(void* const* d_in, const int* in_sizes, int n_in,
                              void* d_out, int out_size, void* d_ws, size_t ws_size,
                              hipStream_t stream) {
  const float* syndrome     = (const float*)d_in[0];
  const float* table_keys   = (const float*)d_in[1];
  const float* table_values = (const float*)d_in[2];

  const int Bn = in_sizes[0] / L_DIM;
  const int Tn = in_sizes[1] / L_DIM;

  uint8_t* ws = (uint8_t*)d_ws;
  size_t off = 0;
  unsigned char* Apacked = (unsigned char*)(ws + off); off += (size_t)Bn * L_DIM;
  unsigned char* Kpacked = (unsigned char*)(ws + off); off += (size_t)Tn * L_DIM;
  off = (off + 255) & ~(size_t)255;
  int* sumA  = (int*)(ws + off); off += (size_t)Bn * sizeof(int);
  int* sumK  = (int*)(ws + off); off += (size_t)Tn * sizeof(int);
  int* match = (int*)(ws + off); off += (size_t)Bn * sizeof(int);
  (void)ws_size; (void)n_in; (void)out_size;

  pack_rows<<<Bn, L_DIM, 0, stream>>>(syndrome, Apacked, sumA);
  pack_rows<<<Tn, L_DIM, 0, stream>>>(table_keys, Kpacked, sumK);

  init_match<<<(Bn + 255) / 256, 256, 0, stream>>>(match, Bn, Tn);

  const int nTileM = Bn / 16;
  const int nTileT = Tn / 16;
  const int totalWaves = nTileM * TCHUNKS;
  const int blocks = (totalWaves + 7) / 8;
  match_gemm<<<blocks, 256, 0, stream>>>(Apacked, Kpacked, sumA, sumK, match,
                                         nTileM, nTileT, Tn);

  gather_out<<<Bn, L_DIM, 0, stream>>>(match, table_values, (float*)d_out, Tn);
}
